// Attention_75522704933073
// MI455X (gfx1250) — matrix operand order repaired and checked by reading, not compiled, not run on hardware
//
#include <hip/hip_runtime.h>

typedef __attribute__((ext_vector_type(16))) _Float16 v16h;
typedef __attribute__((ext_vector_type(8)))  _Float16 v8h;
typedef __attribute__((ext_vector_type(4)))  _Float16 v4h;
typedef __attribute__((ext_vector_type(8)))  float    v8f;

#define NUM_HEADS 16
#define NUM_KV    4
#define HEAD_DIM  128
#define WINDOW    1024
#define SCALE     0.08838834764831845f
#define LOG2E     1.4426950408889634f
#define SCALE2    (SCALE * LOG2E)          // logits straight into log2 domain
#define SEQ       2048
#define QROW      (NUM_HEADS * HEAD_DIM)   // 2048 floats
#define KVROW     (NUM_KV * HEAD_DIM)      // 512 floats

__global__ __launch_bounds__(128) void fa_swa_sink_kernel(
    const float* __restrict__ q, const float* __restrict__ k,
    const float* __restrict__ v, const float* __restrict__ sinks,
    float* __restrict__ out)
{
    __shared__ _Float16 Ksh[32 * 128];    // [key][d]  row-major f16
    __shared__ _Float16 Vsh[128 * 32];    // [d][key]  transposed f16
    __shared__ _Float16 Psh[4 * 16 * 32]; // per-wave P tile [m][kk]

    const int tid  = threadIdx.x;
    const int lane = tid & 31;
    const int wave = tid >> 5;
    const int lg   = lane >> 4;   // 16-lane group 0/1
    const int ln   = lane & 15;

    const int QB  = SEQ / 64;
    int bid = blockIdx.x;
    const int qb  = bid % QB;
    const int h   = (bid / QB) % NUM_HEADS;
    const int b   = bid / (QB * NUM_HEADS);
    const int kvh = h / (NUM_HEADS / NUM_KV);

    const int q0   = qb * 64;
    const int qw   = q0 + wave * 16;   // this wave's first query row
    const int seqb = b * SEQ;

    // staging decomposition: thread owns keys [4g,4g+3] x d [8e,8e+7]
    const int sg  = tid & 7;          // key group
    const int se  = tid >> 3;         // d group (0..15)
    const int dof = se * 8;

    // ---- Load Q fragments (A-layout), f32 -> f16, kept in registers ----
    v16h qa[4];
    {
        const float* qp = q + (size_t)(seqb + qw + ln) * QROW + (size_t)h * HEAD_DIM;
#pragma unroll
        for (int c = 0; c < 4; ++c) {
            int off = c * 32 + lg * 16;
            float4 a0 = *(const float4*)(qp + off);
            float4 a1 = *(const float4*)(qp + off + 4);
            float4 b0 = *(const float4*)(qp + off + 8);
            float4 b1 = *(const float4*)(qp + off + 12);
            v16h t;
            t[0]=(_Float16)a0.x; t[1]=(_Float16)a0.y; t[2]=(_Float16)a0.z; t[3]=(_Float16)a0.w;
            t[4]=(_Float16)a1.x; t[5]=(_Float16)a1.y; t[6]=(_Float16)a1.z; t[7]=(_Float16)a1.w;
            t[8]=(_Float16)b0.x; t[9]=(_Float16)b0.y; t[10]=(_Float16)b0.z; t[11]=(_Float16)b0.w;
            t[12]=(_Float16)b1.x; t[13]=(_Float16)b1.y; t[14]=(_Float16)b1.z; t[15]=(_Float16)b1.w;
            qa[c] = t;
        }
    }

    // ---- Accumulators & online-softmax state (log2 domain) ----
    v8f acc[8];
#pragma unroll
    for (int t8 = 0; t8 < 8; ++t8) acc[t8] = (v8f){};
    float mi[8], li[8];
#pragma unroll
    for (int r = 0; r < 8; ++r) { mi[r] = -1e30f; li[r] = 0.f; }

    _Float16* Pw = Psh + wave * (16 * 32);

    int kstart = q0 - WINDOW; if (kstart < 0) kstart = 0; kstart &= ~31;
    const int kend = q0 + 63;

    for (int kt = kstart; kt <= kend; kt += 32) {
        // ---- Branchless cooperative staging ----
        // OOB keys (spos >= SEQ) are clamped: they are always causally masked,
        // so the staged values are never used (P == 0 there).
        {
            v8h vreg[4];
#pragma unroll
            for (int kk = 0; kk < 4; ++kk) {
                int key  = 4 * sg + kk;
                int spos = kt + key;
                int sposc = spos < SEQ ? spos : SEQ - 1;
                const float* kp = k + (size_t)(seqb + sposc) * KVROW + kvh * HEAD_DIM + dof;
                const float* vp = v + (size_t)(seqb + sposc) * KVROW + kvh * HEAD_DIM + dof;
                if (kt + 32 <= kend && spos + 32 < SEQ) {
                    __builtin_prefetch(kp + 32 * KVROW, 0, 1);
                    __builtin_prefetch(vp + 32 * KVROW, 0, 1);
                }
                float4 x0 = *(const float4*)(kp);
                float4 x1 = *(const float4*)(kp + 4);
                v8h kh;
                kh[0]=(_Float16)x0.x; kh[1]=(_Float16)x0.y; kh[2]=(_Float16)x0.z; kh[3]=(_Float16)x0.w;
                kh[4]=(_Float16)x1.x; kh[5]=(_Float16)x1.y; kh[6]=(_Float16)x1.z; kh[7]=(_Float16)x1.w;
                *(v8h*)(Ksh + key * 128 + dof) = kh;
                float4 y0 = *(const float4*)(vp);
                float4 y1 = *(const float4*)(vp + 4);
                v8h vh;
                vh[0]=(_Float16)y0.x; vh[1]=(_Float16)y0.y; vh[2]=(_Float16)y0.z; vh[3]=(_Float16)y0.w;
                vh[4]=(_Float16)y1.x; vh[5]=(_Float16)y1.y; vh[6]=(_Float16)y1.z; vh[7]=(_Float16)y1.w;
                vreg[kk] = vh;
            }
            // register transpose: 4 keys per b64 store into Vsh[d][key]
#pragma unroll
            for (int j = 0; j < 8; ++j) {
                v4h t;
                t[0] = vreg[0][j]; t[1] = vreg[1][j];
                t[2] = vreg[2][j]; t[3] = vreg[3][j];
                *(v4h*)(Vsh + (dof + j) * 32 + 4 * sg) = t;
            }
        }
        __syncthreads();

        // wave-uniform activity test (EXEC stays all-ones around WMMA)
        bool active = (kt <= qw + 15) && (kt + 31 >= qw - WINDOW);
        if (active) {
            // ---- S = Q * K^T over two 16-key sub-tiles ----
            v8f s0 = (v8f){};
            v8f s1 = (v8f){};
#pragma unroll
            for (int c = 0; c < 4; ++c) {
                v16h kb = *(const v16h*)(Ksh + ln * 128 + c * 32 + lg * 16);
                s0 = __builtin_amdgcn_wmma_f32_16x16x32_f16(false, qa[c], false, kb,
                                                            (short)0, s0, false, false);
            }
#pragma unroll
            for (int c = 0; c < 4; ++c) {
                v16h kb = *(const v16h*)(Ksh + (16 + ln) * 128 + c * 32 + lg * 16);
                s1 = __builtin_amdgcn_wmma_f32_16x16x32_f16(false, qa[c], false, kb,
                                                            (short)0, s1, false, false);
            }

            // fully-visible tile for ALL rows of this wave? (common case)
            const bool full = (kt + 31 <= qw) && (kt >= qw + 15 - WINDOW);
            const int  j0 = kt + ln, j1 = kt + 16 + ln;

            if (full) {
                // ---- unmasked online softmax (log2 domain) ----
#pragma unroll
                for (int r = 0; r < 8; ++r) {
                    float e0 = s0[r] * SCALE2;
                    float e1 = s1[r] * SCALE2;
                    float tm = fmaxf(e0, e1);
                    tm = fmaxf(tm, __shfl_xor(tm, 1, 32));
                    tm = fmaxf(tm, __shfl_xor(tm, 2, 32));
                    tm = fmaxf(tm, __shfl_xor(tm, 4, 32));
                    tm = fmaxf(tm, __shfl_xor(tm, 8, 32));
                    float mn = fmaxf(mi[r], tm);
                    float al = __builtin_amdgcn_exp2f(mi[r] - mn);
                    float p0 = __builtin_amdgcn_exp2f(e0 - mn);
                    float p1 = __builtin_amdgcn_exp2f(e1 - mn);
                    float rs = p0 + p1;
                    rs += __shfl_xor(rs, 1, 32);
                    rs += __shfl_xor(rs, 2, 32);
                    rs += __shfl_xor(rs, 4, 32);
                    rs += __shfl_xor(rs, 8, 32);
                    li[r] = li[r] * al + rs;
                    mi[r] = mn;
#pragma unroll
                    for (int t8 = 0; t8 < 8; ++t8) acc[t8][r] *= al;
                    int m = r + lg * 8;
                    Pw[m * 32 + ln]      = (_Float16)p0;
                    Pw[m * 32 + 16 + ln] = (_Float16)p1;
                }
            } else {
                // ---- masked path: causal + sliding window ----
#pragma unroll
                for (int r = 0; r < 8; ++r) {
                    int i = qw + lg * 8 + r;
                    bool vis0 = (j0 <= i) && ((i - j0) <= WINDOW);
                    bool vis1 = (j1 <= i) && ((i - j1) <= WINDOW);
                    float e0 = vis0 ? s0[r] * SCALE2 : -1e30f;
                    float e1 = vis1 ? s1[r] * SCALE2 : -1e30f;
                    float tm = fmaxf(e0, e1);
                    tm = fmaxf(tm, __shfl_xor(tm, 1, 32));
                    tm = fmaxf(tm, __shfl_xor(tm, 2, 32));
                    tm = fmaxf(tm, __shfl_xor(tm, 4, 32));
                    tm = fmaxf(tm, __shfl_xor(tm, 8, 32));
                    float mn = fmaxf(mi[r], tm);
                    float al = __builtin_amdgcn_exp2f(mi[r] - mn);
                    float p0 = vis0 ? __builtin_amdgcn_exp2f(e0 - mn) : 0.f;
                    float p1 = vis1 ? __builtin_amdgcn_exp2f(e1 - mn) : 0.f;
                    float rs = p0 + p1;
                    rs += __shfl_xor(rs, 1, 32);
                    rs += __shfl_xor(rs, 2, 32);
                    rs += __shfl_xor(rs, 4, 32);
                    rs += __shfl_xor(rs, 8, 32);
                    li[r] = li[r] * al + rs;
                    mi[r] = mn;
#pragma unroll
                    for (int t8 = 0; t8 < 8; ++t8) acc[t8][r] *= al;
                    int m = r + lg * 8;
                    Pw[m * 32 + ln]      = (_Float16)p0;
                    Pw[m * 32 + 16 + ln] = (_Float16)p1;
                }
            }
            asm volatile("s_wait_dscnt 0" ::: "memory");

            // ---- Reload P in A-layout ----
            v16h pa;
            {
                v8h plo = *(const v8h*)(Pw + ln * 32 + lg * 16);
                v8h phi = *(const v8h*)(Pw + ln * 32 + lg * 16 + 8);
#pragma unroll
                for (int e = 0; e < 8; ++e) { pa[e] = plo[e]; pa[8 + e] = phi[e]; }
            }

            // ---- O += P * V over 8 d-tiles ----
#pragma unroll
            for (int t8 = 0; t8 < 8; ++t8) {
                v16h vb = *(const v16h*)(Vsh + (t8 * 16 + ln) * 32 + lg * 16);
                acc[t8] = __builtin_amdgcn_wmma_f32_16x16x32_f16(false, pa, false, vb,
                                                                 (short)0, acc[t8], false, false);
            }
        }
        __syncthreads();
    }

    // ---- Sink denominator (log2 domain) + writeback ----
    float sk2 = sinks[h] * LOG2E;
    float inv[8];
#pragma unroll
    for (int r = 0; r < 8; ++r)
        inv[r] = 1.f / (li[r] + __builtin_amdgcn_exp2f(sk2 - mi[r]));

#pragma unroll
    for (int t8 = 0; t8 < 8; ++t8) {
#pragma unroll
        for (int r = 0; r < 8; ++r) {
            int m = r + lg * 8;
            out[(size_t)(seqb + qw + m) * QROW + h * HEAD_DIM + t8 * 16 + ln] =
                acc[t8][r] * inv[r];
        }
    }
}

extern "C" void kernel_launch(void* const* d_in, const int* in_sizes, int n_in,
                              void* d_out, int out_size, void* d_ws, size_t ws_size,
                              hipStream_t stream) {
    const float* q     = (const float*)d_in[0];
    const float* k     = (const float*)d_in[1];
    const float* v     = (const float*)d_in[2];
    const float* sinks = (const float*)d_in[3];
    float* out = (float*)d_out;

    int T = in_sizes[0] / QROW;     // total tokens
    int B = T / SEQ;                // batch
    dim3 grid(B * NUM_HEADS * (SEQ / 64));
    dim3 block(128);
    fa_swa_sink_kernel<<<grid, block, 0, stream>>>(q, k, v, sinks, out);
}
